// LoRIMoELinear_91018946937087
// MI455X (gfx1250) — compile-verified, matrix operand order checked by reading, not run on hardware
//
#include <hip/hip_runtime.h>
#include <hip/hip_bf16.h>
#include <stdint.h>

typedef __attribute__((ext_vector_type(16))) __bf16         v16bf;
typedef __attribute__((ext_vector_type(8)))  float          v8f;
typedef __attribute__((ext_vector_type(4)))  float          f32x4;
typedef __attribute__((ext_vector_type(8)))  unsigned short u16x8;
typedef __attribute__((ext_vector_type(4)))  unsigned short u16x4;
typedef __attribute__((ext_vector_type(4)))  unsigned int   u32x4;
typedef __attribute__((ext_vector_type(8)))  int            i32x8;
typedef __attribute__((ext_vector_type(4)))  int            i32x4;

#define D_MODEL 2048
#define RANK    32
#define NEXP    8
#define MTOT    16384      // BATCH * SEQ
#define K2TOT   256        // NEXP * RANK

#define BM   128
#define BN   128
#define BK   32
#define LDSS 40            // padded LDS row stride (halfs): 20-bank stride -> conflict-free b128

// LDS arena (halfs): A hi/lo (single-buffered, VALU-converted) + B hi/lo x2 (TDM double-buffered)
#define OFF_AHI   0
#define OFF_ALO   (BM * LDSS)
#define OFF_B0HI  (2 * BM * LDSS)
#define OFF_B0LO  (2 * BM * LDSS + BN * LDSS)
#define OFF_B1HI  (2 * BM * LDSS + 2 * BN * LDSS)
#define OFF_B1LO  (2 * BM * LDSS + 3 * BN * LDSS)
#define SMEM_HALFS (2 * BM * LDSS + 4 * BN * LDSS)   // 30720 halfs = 60 KB

struct FragBits { u16x8 c0, c1; };

__device__ __forceinline__ v16bf load_frag(const unsigned short* s, int row, int laneHalf) {
  const int c0 = laneHalf * 8;
  FragBits fb;
  fb.c0 = *(const u16x8*)(s + row * LDSS + c0);        // K = c0 .. c0+7    -> VGPR 0..3
  fb.c1 = *(const u16x8*)(s + row * LDSS + c0 + 16);   // K = c0+16..c0+23  -> VGPR 4..7
  return __builtin_bit_cast(v16bf, fb);
}

__device__ __forceinline__ void split_one(float x, unsigned short& h, unsigned short& l) {
  const unsigned u = __builtin_bit_cast(unsigned, x);
  h = (unsigned short)(u >> 16);                               // truncated bf16: exact hi part
  const float hf  = __builtin_bit_cast(float, u & 0xffff0000u);
  const float lof = x - hf;                                    // exact residual
  l = (unsigned short)(__builtin_bit_cast(unsigned, lof) >> 16);
}

__device__ __forceinline__ void split_store(unsigned short* shi, unsigned short* slo,
                                            int row, int col, f32x4 v) {
  u16x4 h, l;
#pragma unroll
  for (int i = 0; i < 4; ++i) { unsigned short hh, ll; split_one(v[i], hh, ll); h[i] = hh; l[i] = ll; }
  *(u16x4*)(shi + row * LDSS + col) = h;
  *(u16x4*)(slo + row * LDSS + col) = l;
}

// --- Tensor Data Mover: DMA one [BN rows x BK halfs] bf16 tile into LDS, with
// --- hardware row padding: 16 DWORDs data (32 halfs) then 4 DWORDs pad -> LDSS=40.
__device__ __forceinline__ void tdm_load_tile(const unsigned short* gsrc,
                                              unsigned ldsByteAddr,
                                              unsigned strideElems) {
  const unsigned long long ga = (unsigned long long)(size_t)gsrc;
  u32x4 g0;
  g0[0] = 1u;                                                  // count=1 (valid), no gather
  g0[1] = ldsByteAddr;                                         // lds_addr
  g0[2] = (unsigned)ga;                                        // global_addr[31:0]
  g0[3] = ((unsigned)(ga >> 32) & 0x1ffffffu) | (2u << 30);    // global_addr[56:32] | type=2
  i32x8 g1;
  g1[0] = (1 << 16)        // data_size = 2B
        | (1 << 20)        // pad_enable
        | (3 << 22)        // pad_interval: 16 DWORDs
        | (3 << 25);       // pad_amount:   4 DWORDs
  g1[1] = 0;                                  // abar=0, tensor_dim0 lo16 (dim0 = 1<<20)
  g1[2] = 16;                                 // tensor_dim0 hi16, tensor_dim1 lo16 (dim1 = 1<<20)
  g1[3] = 16 | (BK << 16);                    // tensor_dim1 hi16, tile_dim0 = 32 halfs
  g1[4] = BN;                                 // tile_dim1 = 128 rows, tile_dim2 = 0
  g1[5] = (int)strideElems;                   // tensor_dim0_stride lo32
  g1[6] = 0; g1[7] = 0;
  const i32x4 g2 = {0, 0, 0, 0};
  const i32x4 g3 = {0, 0, 0, 0};
#if defined(__clang_major__) && __clang_major__ >= 23
  const i32x8 g4 = {0, 0, 0, 0, 0, 0, 0, 0};
  __builtin_amdgcn_tensor_load_to_lds(g0, g1, g2, g3, g4, 0);
#else
  __builtin_amdgcn_tensor_load_to_lds(g0, g1, g2, g3, 0);
#endif
}

// ---------------------------------------------------------------------------
// Fused: out[m,o] = b[o] + sum_k X[m,k]*W[o,k] (K=2048) + sum_j C2[m,j]*Am2[o,j] (K=256)
// Split-bf16 WMMA (hiHi + hiLo + loHi). A-side converted in-register (must touch
// VALU anyway); B-side pre-split in ws and DMA'd by the TDM, double-buffered on
// TENSORcnt. All loop-variant selects are uniform scalar selects (no memory
// tables -> no kmcnt stalls in the inner loop).
// ---------------------------------------------------------------------------
__global__ __launch_bounds__(256)
void fused_gemm_kernel(const float* __restrict__ X,            // [MTOT][D_MODEL] f32
                       const float* __restrict__ bias,         // [D_MODEL]
                       const float* __restrict__ C2,           // [MTOT][K2TOT]   f32
                       const unsigned short* __restrict__ Whi, // [D_MODEL][D_MODEL] bf16
                       const unsigned short* __restrict__ Wlo,
                       const unsigned short* __restrict__ Amhi,// [D_MODEL][K2TOT] bf16
                       const unsigned short* __restrict__ Amlo,
                       float* __restrict__ out)                // [MTOT][D_MODEL]
{
  __shared__ __align__(16) unsigned short smem[SMEM_HALFS];

  const int tid  = threadIdx.x;
  const int lane = tid & 31;
  const int wave = tid >> 5;
  const int wm = wave >> 2;           // 0..1  -> 64-row slab
  const int wn = wave & 3;            // 0..3  -> 32-col slab
  const int laneHalf = lane >> 4;
  const int laneMod  = lane & 15;

  const int mBase = blockIdx.y * BM;
  const int nBase = blockIdx.x * BN;

  const int tr = tid >> 3;            // loader row 0..31 (stepped by 32)
  const int tc = (tid & 7) << 2;      // loader col (f32) 0,4,...,28

  const unsigned ldsBase = (unsigned)(size_t)&smem[0];

  v8f acc[4][2];
  const v8f vzero = {0.f, 0.f, 0.f, 0.f, 0.f, 0.f, 0.f, 0.f};
#pragma unroll
  for (int i = 0; i < 4; ++i)
#pragma unroll
    for (int j = 0; j < 2; ++j) acc[i][j] = vzero;

#pragma unroll 1
  for (int phase = 0; phase < 2; ++phase) {
    const float*          Ap     = phase ? C2   : X;
    const unsigned short* BhG    = phase ? Amhi : Whi;
    const unsigned short* BlG    = phase ? Amlo : Wlo;
    const int             ld     = phase ? K2TOT : D_MODEL;
    const int             nk     = phase ? (K2TOT / BK) : (D_MODEL / BK);

    const unsigned short* bhBase = BhG + (size_t)nBase * ld;
    const unsigned short* blBase = BlG + (size_t)nBase * ld;

    if (wave == 0) {  // prologue: DMA first B tile pair into buffer 0
      tdm_load_tile(bhBase, ldsBase + OFF_B0HI * 2, (unsigned)ld);
      tdm_load_tile(blBase, ldsBase + OFF_B0LO * 2, (unsigned)ld);
    }

#pragma unroll 1
    for (int kt = 0; kt < nk; ++kt) {
      const int cur = kt & 1;
      const int kBase = kt * BK;
      // uniform scalar selects (s_cselect), not memory tables:
      const unsigned oBhiCur = cur ? OFF_B1HI : OFF_B0HI;
      const unsigned oBloCur = cur ? OFF_B1LO : OFF_B0LO;
      const unsigned oBhiNxt = cur ? OFF_B0HI : OFF_B1HI;
      const unsigned oBloNxt = cur ? OFF_B0LO : OFF_B1LO;

      // prefetch next B tile pair into the other buffer (its readers finished
      // before the trailing barrier of iteration kt-1)
      if (wave == 0 && (kt + 1) < nk) {
        tdm_load_tile(bhBase + (kBase + BK), ldsBase + oBhiNxt * 2, (unsigned)ld);
        tdm_load_tile(blBase + (kBase + BK), ldsBase + oBloNxt * 2, (unsigned)ld);
      }

      // stage A tile: f32 -> split bf16 hi/lo (VALU conversion is unavoidable here)
#pragma unroll
      for (int rr = 0; rr < BM; rr += 32) {
        const int row = tr + rr;
        const f32x4 a4 = *(const f32x4*)(Ap + (size_t)(mBase + row) * ld + kBase + tc);
        split_store(smem + OFF_AHI, smem + OFF_ALO, row, tc, a4);
      }

      // retire the DMA for the CURRENT buffer (in-order TENSORcnt: <=2 leaves
      // only the just-issued prefetch pair outstanding)
      if (wave == 0) {
        if ((kt + 1) < nk) __builtin_amdgcn_s_wait_tensorcnt(2);
        else               __builtin_amdgcn_s_wait_tensorcnt(0);
      }
      __syncthreads();

      v16bf ah[4], al[4], bh[2], bl[2];
#pragma unroll
      for (int mt = 0; mt < 4; ++mt) {
        const int r = wm * 64 + mt * 16 + laneMod;
        ah[mt] = load_frag(smem + OFF_AHI, r, laneHalf);
        al[mt] = load_frag(smem + OFF_ALO, r, laneHalf);
      }
#pragma unroll
      for (int nt = 0; nt < 2; ++nt) {
        const int r = wn * 32 + nt * 16 + laneMod;
        bh[nt] = load_frag(smem + oBhiCur, r, laneHalf);
        bl[nt] = load_frag(smem + oBloCur, r, laneHalf);
      }
#pragma unroll
      for (int mt = 0; mt < 4; ++mt)
#pragma unroll
        for (int nt = 0; nt < 2; ++nt) {
          acc[mt][nt] = __builtin_amdgcn_wmma_f32_16x16x32_bf16(
              false, ah[mt], false, bh[nt], (short)0, acc[mt][nt], false, false);
          acc[mt][nt] = __builtin_amdgcn_wmma_f32_16x16x32_bf16(
              false, ah[mt], false, bl[nt], (short)0, acc[mt][nt], false, false);
          acc[mt][nt] = __builtin_amdgcn_wmma_f32_16x16x32_bf16(
              false, al[mt], false, bh[nt], (short)0, acc[mt][nt], false, false);
        }
      __syncthreads();
    }
  }

  // Epilogue: C/D layout -> VGPR i: lanes0-15 M=i, lanes16-31 M=8+i; N=laneMod.
#pragma unroll
  for (int nt = 0; nt < 2; ++nt) {
    const int col = nBase + wn * 32 + nt * 16 + laneMod;
    const float bv = bias[col];
#pragma unroll
    for (int mt = 0; mt < 4; ++mt) {
#pragma unroll
      for (int i = 0; i < 8; ++i) {
        const int rowg = mBase + wm * 64 + mt * 16 + laneHalf * 8 + i;
        out[(size_t)rowg * D_MODEL + col] = acc[mt][nt][i] + bv;
      }
    }
  }
}

// ---------------------------------------------------------------------------
// x_proj[m,r] = sum_k x[m,k]*shared_B[r,k] — bandwidth-bound (re-reads x 134 MB),
// AI ~16 FLOP/B << machine balance, so VALU f32 + LDS cross-lane reduction.
// ---------------------------------------------------------------------------
__global__ __launch_bounds__(256)
void xproj_kernel(const float* __restrict__ x, const float* __restrict__ sB,
                  float* __restrict__ xp) {
  __shared__ float red[8][RANK][33];
  const int w    = threadIdx.x >> 5;
  const int lane = threadIdx.x & 31;
  const int m    = blockIdx.x * 8 + w;

  const float* xrow = x + (size_t)m * D_MODEL;
  float acc[RANK];
#pragma unroll
  for (int r = 0; r < RANK; ++r) acc[r] = 0.f;

  for (int k0 = 0; k0 < D_MODEL; k0 += 32) {
    const float xv = xrow[k0 + lane];
#pragma unroll
    for (int r = 0; r < RANK; ++r)
      acc[r] = fmaf(xv, sB[r * D_MODEL + k0 + lane], acc[r]);
  }
#pragma unroll
  for (int r = 0; r < RANK; ++r) red[w][r][lane] = acc[r];
  __syncthreads();
  float s = 0.f;
#pragma unroll
  for (int j = 0; j < 32; ++j) s += red[w][lane][j];
  xp[(size_t)m * RANK + lane] = s;
}

// c[m, e*32+r] = rw[m,e] * xp[m,r]   (kept f32; it is the M-side operand)
__global__ __launch_bounds__(256)
void cmul_kernel(const float* __restrict__ rw, const float* __restrict__ xp,
                 float* __restrict__ c) {
  const int m = blockIdx.x;
  const int j = threadIdx.x;
  const int e = j >> 5, r = j & 31;
  c[(size_t)m * K2TOT + j] = rw[m * NEXP + e] * xp[(size_t)m * RANK + r];
}

// Am2{hi,lo}[o, e*32+r] = split(A[e,o,r] * mask[e,o,r] * es[e])
__global__ __launch_bounds__(256)
void am2_prep_kernel(const float* __restrict__ A, const float* __restrict__ mask,
                     const float* __restrict__ es,
                     unsigned short* __restrict__ Amhi, unsigned short* __restrict__ Amlo) {
  const int idx = blockIdx.x * 256 + threadIdx.x;   // flat [e][o][r]
  const int e = idx >> 16;
  const int o = (idx >> 5) & 2047;
  const int r = idx & 31;
  const float v = A[idx] * mask[idx] * es[e];
  unsigned short h, l;
  split_one(v, h, l);
  Amhi[o * K2TOT + e * 32 + r] = h;
  Amlo[o * K2TOT + e * 32 + r] = l;
}

// W -> split bf16 hi/lo, done ONCE (instead of once per M-block in the GEMM)
__global__ __launch_bounds__(256)
void splitw_kernel(const float* __restrict__ W,
                   unsigned short* __restrict__ Whi, unsigned short* __restrict__ Wlo) {
  const size_t i4 = ((size_t)blockIdx.x * 256 + threadIdx.x) * 4;
  const f32x4 v = *(const f32x4*)(W + i4);
  u16x4 h, l;
#pragma unroll
  for (int i = 0; i < 4; ++i) { unsigned short hh, ll; split_one(v[i], hh, ll); h[i] = hh; l[i] = ll; }
  *(u16x4*)(Whi + i4) = h;
  *(u16x4*)(Wlo + i4) = l;
}

extern "C" void kernel_launch(void* const* d_in, const int* in_sizes, int n_in,
                              void* d_out, int out_size, void* d_ws, size_t ws_size,
                              hipStream_t stream) {
  const float* x    = (const float*)d_in[0];
  const float* rw   = (const float*)d_in[1];
  const float* W    = (const float*)d_in[2];
  const float* b    = (const float*)d_in[3];
  const float* sB   = (const float*)d_in[4];
  const float* A    = (const float*)d_in[5];
  const float* mask = (const float*)d_in[6];
  const float* es   = (const float*)d_in[7];
  float* out = (float*)d_out;

  char* ws = (char*)d_ws;
  float*          xp   = (float*)(ws);                                  //  2 MB
  float*          c    = (float*)(ws + (size_t)(2  << 20));             // 16 MB
  unsigned short* Whi  = (unsigned short*)(ws + (size_t)(18 << 20));    //  8 MB
  unsigned short* Wlo  = (unsigned short*)(ws + (size_t)(26 << 20));    //  8 MB
  unsigned short* Amhi = (unsigned short*)(ws + (size_t)(34 << 20));    //  1 MB
  unsigned short* Amlo = (unsigned short*)(ws + (size_t)(35 << 20));    //  1 MB

  splitw_kernel  <<<(D_MODEL * D_MODEL) / (256 * 4), 256, 0, stream>>>(W, Whi, Wlo);
  am2_prep_kernel<<<(NEXP * D_MODEL * RANK) / 256, 256, 0, stream>>>(A, mask, es, Amhi, Amlo);
  xproj_kernel   <<<MTOT / 8, 256, 0, stream>>>(x, sB, xp);
  cmul_kernel    <<<MTOT, 256, 0, stream>>>(rw, xp, c);

  dim3 grid(D_MODEL / BN, MTOT / BM);
  fused_gemm_kernel<<<grid, 256, 0, stream>>>(x, b, c, Whi, Wlo, Amhi, Amlo, out);
}
